// AttentionBlock_89352499626200
// MI455X (gfx1250) — matrix operand order repaired and checked by reading, not compiled, not run on hardware
//
#include <hip/hip_runtime.h>
#include <hip/hip_bf16.h>

typedef __attribute__((ext_vector_type(16))) _Float16 v16h;
typedef __attribute__((ext_vector_type(8)))  _Float16 v8h;
typedef __attribute__((ext_vector_type(8)))  float    v8f;

#define NB 4
#define NC 512
#define NN 4096          // 64*64 spatial positions
#define NG 8
#define GN_EPS 1e-5f

// padded LDS row strides (f16 units) to spread the 64 banks
#define KSTR 520         // ldsK row stride (512 + 8)
#define VSTR 40          // ldsV row stride (32 + 8)
#define XSTR 40          // gemm ldsX row stride
#define WSTR 40          // gemm ldsW row stride
#define PSTR 40          // P tile row stride
#define TSTR 72          // transpose tile row stride (64 + 8)

union AFrag { v16h v; v8h h[2]; _Float16 e[16]; };
union Acc   { v8f  v; float f[8]; };
union H8    { v8h  v; _Float16 e[8]; };

static __device__ __forceinline__ v8f wmma_f16(const v16h& a, const v16h& b, const v8f& c) {
  // D = A(16x32 f16) * B(32x16 f16) + C(16x16 f32)
  return __builtin_amdgcn_wmma_f32_16x16x32_f16(false, a, false, b, (short)0, c, false, false);
}

// ---------------------------------------------------------------------------
// GroupNorm pass 1: per (batch, group) mean / rstd over 64ch * 4096 px
// ---------------------------------------------------------------------------
__global__ __launch_bounds__(256) void gn_stats_kernel(const float* __restrict__ x,
                                                       float* __restrict__ stats) {
  const int bg = blockIdx.x;              // 0..31
  const int b = bg >> 3, g = bg & 7;
  const float* p = x + ((size_t)b * NC + (size_t)g * 64) * NN;
  const int total = 64 * NN;              // 262144
  float s = 0.f, ss = 0.f;
  for (int i = threadIdx.x; i < total; i += 256) {
    float v = p[i];
    s += v; ss += v * v;
  }
  #pragma unroll
  for (int m = 16; m >= 1; m >>= 1) {
    s  += __shfl_xor(s,  m, 32);
    ss += __shfl_xor(ss, m, 32);
  }
  __shared__ float red[2][8];
  const int wave = threadIdx.x >> 5, lane = threadIdx.x & 31;
  if (lane == 0) { red[0][wave] = s; red[1][wave] = ss; }
  __syncthreads();
  if (threadIdx.x == 0) {
    float ts = 0.f, tss = 0.f;
    #pragma unroll
    for (int w = 0; w < 8; ++w) { ts += red[0][w]; tss += red[1][w]; }
    float mean = ts / (float)total;
    float var  = tss / (float)total - mean * mean;
    stats[bg * 2 + 0] = mean;
    stats[bg * 2 + 1] = rsqrtf(var + GN_EPS);
  }
}

// ---------------------------------------------------------------------------
// GroupNorm pass 2 (fused transpose): x [b][c][n] fp32 -> hn [b][n][c] f16.
// 64x64 tile per block through padded LDS: coalesced reads AND writes.
// ---------------------------------------------------------------------------
__global__ __launch_bounds__(256) void gn_apply_t_kernel(const float* __restrict__ x,
                                                         const float* __restrict__ gamma,
                                                         const float* __restrict__ beta,
                                                         const float* __restrict__ stats,
                                                         _Float16* __restrict__ hn) {
  __shared__ _Float16 tile[64 * TSTR];
  const int tid = threadIdx.x;
  const int n0 = blockIdx.x * 64;
  const int c0 = blockIdx.y * 64;      // == group * 64
  const int b  = blockIdx.z;
  const int bg = b * NG + blockIdx.y;
  const float mean = stats[bg * 2 + 0], rstd = stats[bg * 2 + 1];

  { // read side: 64 c-rows x 64 n, coalesced fp32 reads
    const int cl   = tid >> 2;          // 0..63
    const int nseg = (tid & 3) * 16;
    const int c = c0 + cl;
    const float gm = gamma[c] * rstd;
    const float bt = beta[c] - mean * gm;
    const float* sp = x + ((size_t)b * NC + c) * NN + n0 + nseg;
    H8 t0, t1;
    #pragma unroll
    for (int j = 0; j < 8; ++j) {
      t0.e[j] = (_Float16)(sp[j] * gm + bt);
      t1.e[j] = (_Float16)(sp[8 + j] * gm + bt);
    }
    *(v8h*)(tile + cl * TSTR + nseg)     = t0.v;
    *(v8h*)(tile + cl * TSTR + nseg + 8) = t1.v;
  }
  __syncthreads();
  { // write side: 64 n-rows x 64 c, coalesced f16 writes
    const int nl   = tid >> 2;          // 0..63
    const int cseg = (tid & 3) * 16;
    H8 t0, t1;
    #pragma unroll
    for (int j = 0; j < 8; ++j) {
      t0.e[j] = tile[(cseg + j) * TSTR + nl];
      t1.e[j] = tile[(cseg + 8 + j) * TSTR + nl];
    }
    _Float16* dp = hn + ((size_t)b * NN + n0 + nl) * NC + c0 + cseg;
    *(v8h*)dp       = t0.v;
    *(v8h*)(dp + 8) = t1.v;
  }
}

// ---------------------------------------------------------------------------
// WMMA GEMM: out[d,n] = sum_c W[d,c] * X[n,c] + bias[d]   (per batch z)
// X is ALWAYS [b][n][c] f16. Register-staged pipeline: the global loads for
// K-step cc+32 are issued before the WMMAs of step cc, hiding HBM/L2 latency.
//   MODE 0: store f16 [b][d][n]              (for V)
//   MODE 1: store f16 [b][n][d] (transposed, for Q and K)
//   MODE 2: store f32 [b][d][n] + residual   (proj)
// ---------------------------------------------------------------------------
template <int MODE>
__global__ __launch_bounds__(256) void gemm512_kernel(const float* __restrict__ W,
                                                      const _Float16* __restrict__ X,
                                                      const float* __restrict__ bias,
                                                      const float* __restrict__ resid,
                                                      _Float16* __restrict__ outH,
                                                      float* __restrict__ outF) {
  __shared__ _Float16 ldsW[64 * WSTR];    // [d_local][c_local]
  __shared__ _Float16 ldsX[128 * XSTR];   // [n_local][c_local]

  const int tid = threadIdx.x;
  const int lane = tid & 31;
  const int waveId = tid >> 5;
  const int wd = waveId & 1;            // d half (0..1)
  const int wn = waveId >> 1;           // n quarter (0..3)
  const int n0 = blockIdx.x * 128;
  const int d0 = blockIdx.y * 64;
  const int b  = blockIdx.z;
  const size_t bOff = (size_t)b * NC * NN;
  const int col = lane & 15;
  const int kh  = lane >> 4;

  // staging coordinates
  const int wrow  = tid >> 2;           // 0..63
  const int wcoff = (tid & 3) * 8;
  const int xn    = tid >> 1;           // 0..127
  const int xcseg = (tid & 1) * 16;

  float wreg[8];
  v8h   xreg0, xreg1;
  auto loadT = [&](int cc) {
    const float* wp = W + (size_t)(d0 + wrow) * NC + cc + wcoff;
    #pragma unroll
    for (int j = 0; j < 8; ++j) wreg[j] = wp[j];
    const v8h* sp = (const v8h*)(X + bOff + (size_t)(n0 + xn) * NC + cc + xcseg);
    xreg0 = sp[0]; xreg1 = sp[1];
  };
  auto storeT = [&]() {
    H8 pk;
    #pragma unroll
    for (int j = 0; j < 8; ++j) pk.e[j] = (_Float16)wreg[j];
    *(v8h*)(ldsW + wrow * WSTR + wcoff) = pk.v;
    v8h* dp = (v8h*)(ldsX + xn * XSTR + xcseg);
    dp[0] = xreg0; dp[1] = xreg1;
  };

  Acc acc[2][2];
  #pragma unroll
  for (int i = 0; i < 2; ++i)
    #pragma unroll
    for (int j = 0; j < 2; ++j)
      #pragma unroll
      for (int r = 0; r < 8; ++r) acc[i][j].f[r] = 0.f;

  loadT(0);
  for (int cc = 0; cc < NC; cc += 32) {
    __syncthreads();                    // everyone done reading previous tile
    storeT();
    __syncthreads();                    // tile visible
    if (cc + 32 < NC) loadT(cc + 32);   // issue next tile's global loads now

    AFrag a[2], bf[2];
    #pragma unroll
    for (int ti = 0; ti < 2; ++ti) {
      const int r = wd * 32 + ti * 16 + col;
      a[ti].h[0] = *(const v8h*)(ldsW + r * WSTR + kh * 8);
      a[ti].h[1] = *(const v8h*)(ldsW + r * WSTR + kh * 8 + 16);
    }
    #pragma unroll
    for (int tj = 0; tj < 2; ++tj) {
      const int r = wn * 32 + tj * 16 + col;
      bf[tj].h[0] = *(const v8h*)(ldsX + r * XSTR + kh * 8);
      bf[tj].h[1] = *(const v8h*)(ldsX + r * XSTR + kh * 8 + 16);
    }
    #pragma unroll
    for (int ti = 0; ti < 2; ++ti)
      #pragma unroll
      for (int tj = 0; tj < 2; ++tj)
        acc[ti][tj].v = wmma_f16(a[ti].v, bf[tj].v, acc[ti][tj].v);
  }

  // Epilogue. D-tile mapping: lane<16 -> (M=r, N=col); lane>=16 -> (M=r+8, N=col)
  #pragma unroll
  for (int ti = 0; ti < 2; ++ti) {
    const int dbase = d0 + wd * 32 + ti * 16 + kh * 8;
    #pragma unroll
    for (int tj = 0; tj < 2; ++tj) {
      const int nIdx = n0 + wn * 32 + tj * 16 + col;
      if (MODE == 1) {
        H8 pk;
        #pragma unroll
        for (int r = 0; r < 8; ++r)
          pk.e[r] = (_Float16)(acc[ti][tj].f[r] + bias[dbase + r]);
        *(v8h*)(outH + bOff + (size_t)nIdx * NC + dbase) = pk.v;
      } else if (MODE == 0) {
        #pragma unroll
        for (int r = 0; r < 8; ++r) {
          const int d = dbase + r;
          outH[bOff + (size_t)d * NN + nIdx] = (_Float16)(acc[ti][tj].f[r] + bias[d]);
        }
      } else {
        #pragma unroll
        for (int r = 0; r < 8; ++r) {
          const int d = dbase + r;
          const size_t idx = bOff + (size_t)d * NN + nIdx;
          outF[idx] = acc[ti][tj].f[r] + bias[d] + resid[idx];
        }
      }
    }
  }
}

// ---------------------------------------------------------------------------
// Flash attention, register-staged pipeline:
//  - global loads of K/V tile i+1 issue before the 64 WMMAs of tile i
//  - fragment loads run one step ahead of their consuming WMMA
//  - dedicated P region (no mid-step barrier; waves run the S->softmax->O
//    phases independently)
// Block = 8 waves, wave = 16 query rows, O (16x512 f32) in VGPRs.
// ---------------------------------------------------------------------------
__global__ __launch_bounds__(256) void attention_kernel(const _Float16* __restrict__ qT, // [b][n][c]
                                                        const _Float16* __restrict__ kT, // [b][n][c]
                                                        const _Float16* __restrict__ vM, // [b][c][n]
                                                        _Float16* __restrict__ oT) {     // [b][n][c]
  __shared__ _Float16 ldsK[32 * KSTR];        // [m_local][c]  (padded)
  __shared__ _Float16 ldsV[NC * VSTR];        // [c][m_local]  (padded)
  __shared__ _Float16 ldsP[8 * 16 * PSTR];    // per-wave P tiles
  const int tid = threadIdx.x, lane = tid & 31, waveId = tid >> 5;
  const int b = blockIdx.y;
  const int nBase = blockIdx.x * 128 + waveId * 16;
  const size_t bOff = (size_t)b * NC * NN;
  const int col = lane & 15, kh = lane >> 4;
  const float scale = 0.044194173824159216f;   // 512^-0.5

  // staging coordinates + registers
  const int km = tid >> 3, kseg = (tid & 7) * 64;   // K: 32 rows x 1KB
  const int vrow = tid * 2;                          // V: 512 rows x 64B (2 each)
  uint4 kreg[8], vreg0[4], vreg1[4];
  auto loadKV = [&](int m0) {
    const uint4* sk = (const uint4*)(kT + bOff + (size_t)(m0 + km) * NC + kseg);
    #pragma unroll
    for (int j = 0; j < 8; ++j) kreg[j] = sk[j];
    const uint4* sv0 = (const uint4*)(vM + bOff + (size_t)vrow * NN + m0);
    const uint4* sv1 = (const uint4*)(vM + bOff + (size_t)(vrow + 1) * NN + m0);
    #pragma unroll
    for (int j = 0; j < 4; ++j) { vreg0[j] = sv0[j]; vreg1[j] = sv1[j]; }
  };
  auto storeKV = [&]() {
    uint4* dk = (uint4*)(ldsK + km * KSTR + kseg);
    #pragma unroll
    for (int j = 0; j < 8; ++j) dk[j] = kreg[j];
    uint4* dv0 = (uint4*)(ldsV + vrow * VSTR);
    uint4* dv1 = (uint4*)(ldsV + (vrow + 1) * VSTR);
    #pragma unroll
    for (int j = 0; j < 4; ++j) { dv0[j] = vreg0[j]; dv1[j] = vreg1[j]; }
  };

  // q fragments resident in VGPRs: 16 chunks of K=32 (A-matrix layout)
  AFrag qf[16];
  #pragma unroll
  for (int kc = 0; kc < 16; ++kc) {
    const _Float16* qp = qT + bOff + (size_t)(nBase + col) * NC + kc * 32 + kh * 8;
    qf[kc].h[0] = *(const v8h*)qp;
    qf[kc].h[1] = *(const v8h*)(qp + 16);
  }

  Acc o[32];                                   // 16q x 512c accumulator
  #pragma unroll
  for (int ct = 0; ct < 32; ++ct)
    #pragma unroll
    for (int r = 0; r < 8; ++r) o[ct].f[r] = 0.f;

  float mst[8], lst[8];
  #pragma unroll
  for (int r = 0; r < 8; ++r) { mst[r] = -3.0e38f; lst[r] = 0.f; }

  _Float16* ldsPw = ldsP + waveId * (16 * PSTR);

  loadKV(0);
  for (int m0 = 0; m0 < NN; m0 += 32) {
    __syncthreads();                  // all waves done reading previous tile
    storeKV();
    __syncthreads();                  // tile visible to all waves
    if (m0 + 32 < NN) loadKV(m0 + 32);  // next tile's global loads in flight

    // ---- S = q . k (16q x 32m -> two 16x16 D tiles), frag lookahead ----
    Acc s[2];
    #pragma unroll
    for (int tm = 0; tm < 2; ++tm)
      #pragma unroll
      for (int r = 0; r < 8; ++r) s[tm].f[r] = 0.f;

    auto loadKB = [&](AFrag& f0, AFrag& f1, int kc) {
      const _Float16* bp0 = ldsK + col * KSTR + kc * 32 + kh * 8;
      const _Float16* bp1 = ldsK + (16 + col) * KSTR + kc * 32 + kh * 8;
      f0.h[0] = *(const v8h*)bp0; f0.h[1] = *(const v8h*)(bp0 + 16);
      f1.h[0] = *(const v8h*)bp1; f1.h[1] = *(const v8h*)(bp1 + 16);
    };
    AFrag bc0, bc1, bn0, bn1;
    loadKB(bc0, bc1, 0);
    #pragma unroll
    for (int kc = 0; kc < 16; ++kc) {
      if (kc < 15) loadKB(bn0, bn1, kc + 1);
      s[0].v = wmma_f16(qf[kc].v, bc0.v, s[0].v);
      s[1].v = wmma_f16(qf[kc].v, bc1.v, s[1].v);
      bc0 = bn0; bc1 = bn1;
    }

    // ---- online softmax ----
    float nm[8], corr[8];
    #pragma unroll
    for (int r = 0; r < 8; ++r) { s[0].f[r] *= scale; s[1].f[r] *= scale; }
    #pragma unroll
    for (int r = 0; r < 8; ++r) {
      float mx = fmaxf(s[0].f[r], s[1].f[r]);
      mx = fmaxf(mx, __shfl_xor(mx, 1, 32));
      mx = fmaxf(mx, __shfl_xor(mx, 2, 32));
      mx = fmaxf(mx, __shfl_xor(mx, 4, 32));
      mx = fmaxf(mx, __shfl_xor(mx, 8, 32));
      nm[r] = fmaxf(mst[r], mx);
      corr[r] = __expf(mst[r] - nm[r]);
      mst[r] = nm[r];
    }
    #pragma unroll
    for (int r = 0; r < 8; ++r) {
      s[0].f[r] = __expf(s[0].f[r] - nm[r]);
      s[1].f[r] = __expf(s[1].f[r] - nm[r]);
      float sum = s[0].f[r] + s[1].f[r];
      sum += __shfl_xor(sum, 1, 32);
      sum += __shfl_xor(sum, 2, 32);
      sum += __shfl_xor(sum, 4, 32);
      sum += __shfl_xor(sum, 8, 32);
      lst[r] = lst[r] * corr[r] + sum;
    }
    #pragma unroll
    for (int ct = 0; ct < 32; ++ct)
      #pragma unroll
      for (int r = 0; r < 8; ++r) o[ct].f[r] *= corr[r];

    // P: D layout -> A layout via dedicated per-wave LDS slice (wave-private)
    #pragma unroll
    for (int tm = 0; tm < 2; ++tm)
      #pragma unroll
      for (int r = 0; r < 8; ++r)
        ldsPw[(r + kh * 8) * PSTR + tm * 16 + col] = (_Float16)s[tm].f[r];
    asm volatile("s_wait_dscnt 0" ::: "memory");
    AFrag pf;
    pf.h[0] = *(const v8h*)(ldsPw + col * PSTR + kh * 8);
    pf.h[1] = *(const v8h*)(ldsPw + col * PSTR + 16 + kh * 8);

    // ---- O += P . V (32 c-tiles), frag lookahead ----
    auto loadVB = [&](AFrag& f, int ct) {
      const _Float16* bp = ldsV + (ct * 16 + col) * VSTR + kh * 8;
      f.h[0] = *(const v8h*)bp;
      f.h[1] = *(const v8h*)(bp + 16);
    };
    AFrag vcur, vnxt;
    loadVB(vcur, 0);
    #pragma unroll
    for (int ct = 0; ct < 32; ++ct) {
      if (ct < 31) loadVB(vnxt, ct + 1);
      o[ct].v = wmma_f16(pf.v, vcur.v, o[ct].v);
      vcur = vnxt;
    }
  }

  // normalize + store oT[b][n][c]
  #pragma unroll
  for (int r = 0; r < 8; ++r) lst[r] = 1.0f / lst[r];
  #pragma unroll
  for (int ct = 0; ct < 32; ++ct) {
    #pragma unroll
    for (int r = 0; r < 8; ++r) {
      const int row = nBase + r + kh * 8;
      oT[bOff + (size_t)row * NC + ct * 16 + col] = (_Float16)(o[ct].f[r] * lst[r]);
    }
  }
}

// ---------------------------------------------------------------------------
extern "C" void kernel_launch(void* const* d_in, const int* in_sizes, int n_in,
                              void* d_out, int out_size, void* d_ws, size_t ws_size,
                              hipStream_t stream) {
  const float* x     = (const float*)d_in[0];
  const float* gamma = (const float*)d_in[1];
  const float* beta  = (const float*)d_in[2];
  const float* qw    = (const float*)d_in[3];
  const float* qb    = (const float*)d_in[4];
  const float* kw    = (const float*)d_in[5];
  const float* kb    = (const float*)d_in[6];
  const float* vw    = (const float*)d_in[7];
  const float* vb    = (const float*)d_in[8];
  const float* pw    = (const float*)d_in[9];
  const float* pb    = (const float*)d_in[10];
  float* out = (float*)d_out;

  char* ws = (char*)d_ws;
  float* stats  = (float*)ws;
  _Float16* hn  = (_Float16*)(ws + 4096);       // [b][n][c]
  const size_t E = (size_t)NB * NC * NN;
  _Float16* qT = hn + E;                        // [b][n][c]
  _Float16* kT = qT + E;                        // [b][n][c]
  _Float16* vM = kT + E;                        // [b][c][n]
  _Float16* oT = vM + E;                        // [b][n][c]

  gn_stats_kernel<<<dim3(NB * NG), 256, 0, stream>>>(x, stats);
  gn_apply_t_kernel<<<dim3(NN / 64, NC / 64, NB), 256, 0, stream>>>(x, gamma, beta, stats, hn);

  dim3 gg(NN / 128, NC / 64, NB);
  gemm512_kernel<1><<<gg, 256, 0, stream>>>(qw, hn, qb, nullptr, qT, nullptr);
  gemm512_kernel<1><<<gg, 256, 0, stream>>>(kw, hn, kb, nullptr, kT, nullptr);
  gemm512_kernel<0><<<gg, 256, 0, stream>>>(vw, hn, vb, nullptr, vM, nullptr);

  attention_kernel<<<dim3(NN / 128, NB), 256, 0, stream>>>(qT, kT, vM, oT);

  gemm512_kernel<2><<<gg, 256, 0, stream>>>(pw, oT, pb, x, nullptr, out);
}
